// BasicMambaBlock_53738630808035
// MI455X (gfx1250) — matrix operand order repaired and checked by reading, not compiled, not run on hardware
//
#include <hip/hip_runtime.h>
#include <hip/hip_bf16.h>

typedef __attribute__((ext_vector_type(16))) _Float16 v16h;
typedef __attribute__((ext_vector_type(8)))  float    v8f;

#define B_    2
#define L_    1024
#define DM_   1024
#define DI_   2048
#define NST   16
#define DTR_  64
#define MTOT  (B_ * L_)        // 2048
#define XZW   (2 * DI_)        // 4096
#define XDBLW (DTR_ + 2*NST)   // 96

// ---- CDNA5 async global->LDS path (guarded; falls back to sync staging) ----
#if defined(__AMDGCN__) && __has_builtin(__builtin_amdgcn_global_load_async_to_lds_b128) && __has_builtin(__builtin_amdgcn_s_wait_asynccnt)
#define USE_ASYNC 1
typedef int v4i_ __attribute__((ext_vector_type(4)));
typedef __attribute__((address_space(1))) v4i_* gas_ptr;
typedef __attribute__((address_space(3))) v4i_* lds_ptr;
// generic LDS addr: low 32 bits are the LDS offset (flat aperture rule)
#define TO_GAS(p) ((gas_ptr)(unsigned long long)(p))
#define TO_LDS(p) ((lds_ptr)(unsigned int)(unsigned long long)(p))
#define ASYNC_CP16(dst, src) \
    __builtin_amdgcn_global_load_async_to_lds_b128(TO_GAS(src), TO_LDS(dst), 0, 0)
#else
#define USE_ASYNC 0
#endif

// ---------------------------------------------------------------- LayerNorm
__global__ __launch_bounds__(256)
void ln_kernel(const float* __restrict__ x, const float* __restrict__ g,
               const float* __restrict__ bta, _Float16* __restrict__ xn) {
    int m = blockIdx.x;
    const float* row = x + (size_t)m * DM_;
    __shared__ float red[256];
    float v[4];
    float s = 0.f;
#pragma unroll
    for (int i = 0; i < 4; ++i) { v[i] = row[threadIdx.x + i * 256]; s += v[i]; }
    red[threadIdx.x] = s; __syncthreads();
    for (int off = 128; off > 0; off >>= 1) {
        if (threadIdx.x < off) red[threadIdx.x] += red[threadIdx.x + off];
        __syncthreads();
    }
    float mu = red[0] * (1.f / DM_);
    __syncthreads();
    float s2 = 0.f;
#pragma unroll
    for (int i = 0; i < 4; ++i) { float d = v[i] - mu; s2 += d * d; }
    red[threadIdx.x] = s2; __syncthreads();
    for (int off = 128; off > 0; off >>= 1) {
        if (threadIdx.x < off) red[threadIdx.x] += red[threadIdx.x + off];
        __syncthreads();
    }
    float rstd = rsqrtf(red[0] * (1.f / DM_) + 1e-5f);
#pragma unroll
    for (int i = 0; i < 4; ++i) {
        int c = threadIdx.x + i * 256;
        xn[(size_t)m * DM_ + c] = (_Float16)((v[i] - mu) * rstd * g[c] + bta[c]);
    }
}

// ----------------------------------- weight convert (+ zero padding) f32->f16
__global__ __launch_bounds__(256)
void cvt_kernel(const float* __restrict__ src, _Float16* __restrict__ dst,
                int count, int total) {
    int i = blockIdx.x * 256 + threadIdx.x;
    if (i < total) dst[i] = (i < count) ? (_Float16)src[i] : (_Float16)0.f;
}

// ---------------------- WMMA GEMM: C = A(f16, MxK) * B(f16, NtotxK)^T
// Block tile 128(M) x 64(N), 8 waves: wave (wm 0..3, wn 0..1) owns 32x32,
// i.e. 2 A-frags x 2 B-frags -> 4 WMMA per 32-K step.
// EPI: 0 plain | 1 xdbl + dt-f16 split | 2 bias+softplus | 3 residual add
template <int EPI>
__global__ __launch_bounds__(256)
void gemm_kernel(const _Float16* __restrict__ A, int lda,
                 const _Float16* __restrict__ Bh, int ldb,
                 float* __restrict__ C, int ldc,
                 int Ntot, int Ktot,
                 const float* __restrict__ aux0, void* __restrict__ aux1) {
    __shared__ __align__(16) _Float16 As[2][128 * 40];
    __shared__ __align__(16) _Float16 Bs[2][64 * 40];

    const int m0 = blockIdx.x * 128;
    const int n0 = blockIdx.y * 64;
    const int tid  = threadIdx.x;
    const int wave = tid >> 5, lane = tid & 31;
    const int wm = wave & 3, wn = wave >> 2;
    const int mrow = lane & 15, lh = lane >> 4;

    v8f acc[2][2] = {};

    const int ar = tid >> 1, ac = (tid & 1) * 16;  // A copy: row 0..127, chunk 0/16
    const int br = tid >> 2, bc = (tid & 3) * 8;   // B copy: row 0..63, chunk 0..24
    const _Float16* gA = A  + (size_t)(m0 + ar) * lda + ac;
    const _Float16* gB = Bh + (size_t)(n0 + br) * ldb + bc;

    const int nIter = Ktot / 32;

#if USE_ASYNC
    // -------- double-buffered async pipeline: 1 barrier / iteration
    {
        ASYNC_CP16(&As[0][ar * 40 + ac],     gA);
        ASYNC_CP16(&As[0][ar * 40 + ac + 8], gA + 8);
        ASYNC_CP16(&Bs[0][br * 40 + bc],     gB);
    }
    for (int it = 0; it < nIter; ++it) {
        const int cur = it & 1;
        __builtin_amdgcn_s_wait_asynccnt(0);
        __syncthreads();   // tiles resident; previous readers of other buffer done
        if (it + 1 < nIter) {
            const int k0 = (it + 1) * 32, nb = 1 - cur;
            ASYNC_CP16(&As[nb][ar * 40 + ac],     gA + k0);
            ASYNC_CP16(&As[nb][ar * 40 + ac + 8], gA + k0 + 8);
            ASYNC_CP16(&Bs[nb][br * 40 + bc],     gB + k0);
        }
        v16h af[2], bf[2];
#pragma unroll
        for (int mt = 0; mt < 2; ++mt) {
            const _Float16* p = &As[cur][(wm * 32 + mt * 16 + mrow) * 40 + lh * 8];
#pragma unroll
            for (int i = 0; i < 8; ++i) { af[mt][i] = p[i]; af[mt][8 + i] = p[16 + i]; }
        }
#pragma unroll
        for (int nt = 0; nt < 2; ++nt) {
            const _Float16* p = &Bs[cur][(wn * 32 + nt * 16 + mrow) * 40 + lh * 8];
#pragma unroll
            for (int i = 0; i < 8; ++i) { bf[nt][i] = p[i]; bf[nt][8 + i] = p[16 + i]; }
        }
#pragma unroll
        for (int mt = 0; mt < 2; ++mt)
#pragma unroll
            for (int nt = 0; nt < 2; ++nt)
                acc[mt][nt] = __builtin_amdgcn_wmma_f32_16x16x32_f16(
                    false, af[mt], false, bf[nt], (short)0, acc[mt][nt], false, false);
    }
#else
    // -------- synchronous fallback (single buffer)
    for (int it = 0; it < nIter; ++it) {
        const int k0 = it * 32;
        *(float4*)&As[0][ar * 40 + ac]     = *(const float4*)(gA + k0);
        *(float4*)&As[0][ar * 40 + ac + 8] = *(const float4*)(gA + k0 + 8);
        *(float4*)&Bs[0][br * 40 + bc]     = *(const float4*)(gB + k0);
        if (it + 1 < nIter) __builtin_prefetch(gA + k0 + 32, 0, 1);
        __syncthreads();
        v16h af[2], bf[2];
#pragma unroll
        for (int mt = 0; mt < 2; ++mt) {
            const _Float16* p = &As[0][(wm * 32 + mt * 16 + mrow) * 40 + lh * 8];
#pragma unroll
            for (int i = 0; i < 8; ++i) { af[mt][i] = p[i]; af[mt][8 + i] = p[16 + i]; }
        }
#pragma unroll
        for (int nt = 0; nt < 2; ++nt) {
            const _Float16* p = &Bs[0][(wn * 32 + nt * 16 + mrow) * 40 + lh * 8];
#pragma unroll
            for (int i = 0; i < 8; ++i) { bf[nt][i] = p[i]; bf[nt][8 + i] = p[16 + i]; }
        }
#pragma unroll
        for (int mt = 0; mt < 2; ++mt)
#pragma unroll
            for (int nt = 0; nt < 2; ++nt)
                acc[mt][nt] = __builtin_amdgcn_wmma_f32_16x16x32_f16(
                    false, af[mt], false, bf[nt], (short)0, acc[mt][nt], false, false);
        __syncthreads();
    }
#endif

    // -------- epilogue: VGPR j -> row +j+lh*8, col = subtile base + mrow
#pragma unroll
    for (int mt = 0; mt < 2; ++mt) {
#pragma unroll
        for (int nt = 0; nt < 2; ++nt) {
#pragma unroll
            for (int j = 0; j < 8; ++j) {
                const int row = m0 + wm * 32 + mt * 16 + j + lh * 8;
                const int col = n0 + wn * 32 + nt * 16 + mrow;
                float v = acc[mt][nt][j];
                if (EPI == 0) {
                    if (col < Ntot) C[(size_t)row * ldc + col] = v;
                } else if (EPI == 1) {
                    _Float16* dtf = (_Float16*)aux1;
                    if (col < Ntot) {
                        C[(size_t)row * ldc + col] = v;
                        if (col < DTR_) dtf[(size_t)row * DTR_ + col] = (_Float16)v;
                    }
                } else if (EPI == 2) {
                    if (col < Ntot) {
                        float t = v + aux0[col];
                        C[(size_t)row * ldc + col] = (t > 20.f) ? t : log1pf(__expf(t));
                    }
                } else { // EPI == 3
                    if (col < Ntot)
                        C[(size_t)row * ldc + col] = v + aux0[(size_t)row * ldc + col];
                }
            }
        }
    }
}

// ------------------------------------------ depthwise causal conv (K=4) + SiLU
__global__ __launch_bounds__(256)
void conv_silu_kernel(const float* __restrict__ xz, const float* __restrict__ cw,
                      const float* __restrict__ cb,
                      float* __restrict__ u, _Float16* __restrict__ uh) {
    int idx = blockIdx.x * blockDim.x + threadIdx.x;   // over MTOT*DI
    int d = idx & (DI_ - 1);
    int m = idx >> 11;
    int b = m >> 10, l = m & (L_ - 1);
    float acc = cb[d];
#pragma unroll
    for (int j = 0; j < 4; ++j) {
        int ls = l - 3 + j;
        if (ls >= 0) acc += cw[d * 4 + j] * xz[(size_t)(b * L_ + ls) * XZW + d];
    }
    float s = acc / (1.f + __expf(-acc));   // SiLU
    u[idx]  = s;
    uh[idx] = (_Float16)s;
}

// ------------------------------------------------------- selective scan + gate
__global__ __launch_bounds__(256)
void scan_kernel(const float* __restrict__ delta, const float* __restrict__ u,
                 const float* __restrict__ xdbl, const float* __restrict__ xz,
                 const float* __restrict__ A_log, const float* __restrict__ Dv,
                 _Float16* __restrict__ yh) {
    int c = blockIdx.x * blockDim.x + threadIdx.x;   // 0 .. B*DI-1
    int d = c & (DI_ - 1);
    int b = c >> 11;
    float Am[NST], h[NST];
#pragma unroll
    for (int n = 0; n < NST; ++n) { Am[n] = -__expf(A_log[d * NST + n]); h[n] = 0.f; }
    const float Dd = Dv[d];
    for (int l = 0; l < L_; ++l) {
        int m = b * L_ + l;
        float dl = delta[(size_t)m * DI_ + d];
        float uu = u[(size_t)m * DI_ + d];
        const float* xr = xdbl + (size_t)m * XDBLW;   // wave-uniform row
        float y = 0.f;
#pragma unroll
        for (int n = 0; n < NST; ++n) {
            h[n] = __expf(dl * Am[n]) * h[n] + dl * xr[DTR_ + n] * uu;
            y += h[n] * xr[DTR_ + NST + n];
        }
        y += uu * Dd;
        float zz = xz[(size_t)m * XZW + DI_ + d];
        y *= zz / (1.f + __expf(-zz));                // * silu(z)
        yh[(size_t)m * DI_ + d] = (_Float16)y;
    }
}

// ---------------------------------------------------------------------------
extern "C" void kernel_launch(void* const* d_in, const int* in_sizes, int n_in,
                              void* d_out, int out_size, void* d_ws, size_t ws_size,
                              hipStream_t stream) {
    (void)in_sizes; (void)n_in; (void)out_size; (void)ws_size;
    const float* x          = (const float*)d_in[0];
    const float* ln_g       = (const float*)d_in[1];
    const float* ln_b       = (const float*)d_in[2];
    const float* in_proj_w  = (const float*)d_in[3];   // (4096,1024)
    const float* conv_w     = (const float*)d_in[4];   // (2048,1,4)
    const float* conv_b     = (const float*)d_in[5];
    const float* x_proj_w   = (const float*)d_in[6];   // (96,2048)
    const float* dt_proj_w  = (const float*)d_in[7];   // (2048,64)
    const float* dt_proj_b  = (const float*)d_in[8];
    const float* A_log      = (const float*)d_in[9];   // (2048,16)
    const float* Dv         = (const float*)d_in[10];
    const float* out_proj_w = (const float*)d_in[11];  // (1024,2048)
    float* out = (float*)d_out;

    char* ws = (char*)d_ws;
    size_t off = 0;
    auto carve = [&](size_t bytes) { char* p = ws + off; off = (off + bytes + 255) & ~(size_t)255; return p; };
    _Float16* xn_h = (_Float16*)carve((size_t)MTOT * DM_ * 2);
    float*    xz   = (float*)   carve((size_t)MTOT * XZW * 4);
    float*    u    = (float*)   carve((size_t)MTOT * DI_ * 4);
    _Float16* uh   = (_Float16*)carve((size_t)MTOT * DI_ * 2);
    float*    xdbl = (float*)   carve((size_t)MTOT * XDBLW * 4);
    _Float16* dtf  = (_Float16*)carve((size_t)MTOT * DTR_ * 2);
    float*    dlt  = (float*)   carve((size_t)MTOT * DI_ * 4);
    _Float16* yh   = (_Float16*)carve((size_t)MTOT * DI_ * 2);
    _Float16* w1h  = (_Float16*)carve((size_t)XZW * DM_ * 2);   // in_proj f16
    _Float16* xph  = (_Float16*)carve((size_t)128 * DI_ * 2);   // x_proj f16, padded 96->128 rows
    _Float16* dtph = (_Float16*)carve((size_t)DI_ * DTR_ * 2);  // dt_proj f16
    _Float16* oph  = (_Float16*)carve((size_t)DM_ * DI_ * 2);   // out_proj f16

    // 0) weights -> f16 (once; reused by every M-block)
    cvt_kernel<<<(XZW * DM_) / 256, 256, 0, stream>>>(in_proj_w, w1h, XZW * DM_, XZW * DM_);
    cvt_kernel<<<(128 * DI_) / 256, 256, 0, stream>>>(x_proj_w, xph, XDBLW * DI_, 128 * DI_);
    cvt_kernel<<<(DI_ * DTR_) / 256, 256, 0, stream>>>(dt_proj_w, dtph, DI_ * DTR_, DI_ * DTR_);
    cvt_kernel<<<(DM_ * DI_) / 256, 256, 0, stream>>>(out_proj_w, oph, DM_ * DI_, DM_ * DI_);
    // 1) LayerNorm -> f16
    ln_kernel<<<MTOT, 256, 0, stream>>>(x, ln_g, ln_b, xn_h);
    // 2) xz = xn @ in_proj_w^T   (2048x1024 . 1024x4096)
    gemm_kernel<0><<<dim3(MTOT / 128, XZW / 64), 256, 0, stream>>>(
        xn_h, DM_, w1h, DM_, xz, XZW, XZW, DM_, nullptr, nullptr);
    // 3) depthwise causal conv + SiLU -> u (f32 + f16)
    conv_silu_kernel<<<(MTOT * DI_) / 256, 256, 0, stream>>>(xz, conv_w, conv_b, u, uh);
    // 4) xdbl = u @ x_proj_w^T   (2048x2048 . 2048x96), split dt -> f16
    gemm_kernel<1><<<dim3(MTOT / 128, 2), 256, 0, stream>>>(
        uh, DI_, xph, DI_, xdbl, XDBLW, XDBLW, DI_, nullptr, (void*)dtf);
    // 5) delta = softplus(dt @ dt_proj_w^T + b)   (2048x64 . 64x2048)
    gemm_kernel<2><<<dim3(MTOT / 128, DI_ / 64), 256, 0, stream>>>(
        dtf, DTR_, dtph, DTR_, dlt, DI_, DI_, DTR_, dt_proj_b, nullptr);
    // 6) selective scan + D skip + SiLU(z) gate -> y (f16)
    scan_kernel<<<(B_ * DI_) / 256, 256, 0, stream>>>(dlt, u, xdbl, xz, A_log, Dv, yh);
    // 7) out = residual + y @ out_proj_w^T   (2048x2048 . 2048x1024)
    gemm_kernel<3><<<dim3(MTOT / 128, DM_ / 64), 256, 0, stream>>>(
        yh, DI_, oph, DI_, out, DM_, DM_, DI_, x, nullptr);
}
